// DFT_76897094468194
// MI455X (gfx1250) — matrix operand order repaired and checked by reading, not compiled, not run on hardware
//
#include <hip/hip_runtime.h>

// ---------------------------------------------------------------------------
// DFT log-magnitude:  out[b,j] = log( (x@W_re^T)^2 + (x@W_im^T)^2 )
// x: [B,256] f32, W_re/W_im: [256,256] f32 (DFT matrix: symmetric, W^T == W).
// Strategy: f16 WMMA (16x16x32, f32 acc) with hi/lo split-precision:
//   v = v_hi + v_lo (f16 each) ; A@B ~= Ah@Bh + Al@Bh + Ah@Bl  (~22-bit mantissa)
// Block = 8 waves = 128 rows x 256 cols. A fragments resident in VGPRs,
// W staged to LDS as f16 hi/lo in 64-column quarters (144 KB -> 2 blocks/WGP).
// B fragments are software-pipelined (ping-pong) one k-step ahead so LDS
// latency overlaps the WMMA bundle instead of forcing s_wait_dscnt 0.
// ---------------------------------------------------------------------------

typedef __attribute__((ext_vector_type(16))) _Float16 v16h;
typedef __attribute__((ext_vector_type(8)))  _Float16 v8h;
typedef __attribute__((ext_vector_type(4)))  _Float16 v4h;
typedef __attribute__((ext_vector_type(8)))  float    v8f;
typedef __attribute__((ext_vector_type(4)))  float    v4f;

#define NFFT            256
#define NQUART          64                  // output columns staged per LDS pass
#define PITCH           (NQUART + 8)        // 72 halves/row: 144B stride, bank-staggered, 16B aligned
#define ROWS_PER_WAVE   16
#define WAVES           8
#define ROWS_PER_BLOCK  (ROWS_PER_WAVE * WAVES)   // 128
#define KSTEPS          (NFFT / 32)               // 8

union FragB { v16h v; v8h h[2]; };
union FragA { v16h v; _Float16 e[16]; };

static __device__ __forceinline__ v8f wmma_f16(v16h a, v16h b, v8f c) {
  return __builtin_amdgcn_wmma_f32_16x16x32_f16(false, a, false, b, (short)0, c,
                                                false, false);
}

// Load the 4 B fragments (re_hi, re_lo, im_hi, im_lo) for one k-step.
// B layout (16-bit, 32x16): lane&15 = N, halves = K (0-7 -> K 0..7, 8-15 -> K 16..23; lanes 16-31 K +8).
static __device__ __forceinline__ void load_frags(FragB* f,
                                                  const _Float16* wr_hi,
                                                  const _Float16* wr_lo,
                                                  const _Float16* wi_hi,
                                                  const _Float16* wi_lo,
                                                  int lo) {
#pragma unroll
  for (int s = 0; s < 16; ++s) {
    const int o = lo + (s < 8 ? s : s + 8) * PITCH;
    f[0].v[s] = wr_hi[o];
    f[1].v[s] = wr_lo[o];
    f[2].v[s] = wi_hi[o];
    f[3].v[s] = wi_lo[o];
  }
}

__launch_bounds__(256, 1)
__global__ void dft_logmag_kernel(const float* __restrict__ x,
                                  const float* __restrict__ dreal,
                                  const float* __restrict__ dimag,
                                  float* __restrict__ out,
                                  int batch) {
  extern __shared__ __align__(32) _Float16 lds[];
  _Float16* wr_hi = lds;
  _Float16* wr_lo = lds + 1 * NFFT * PITCH;
  _Float16* wi_hi = lds + 2 * NFFT * PITCH;
  _Float16* wi_lo = lds + 3 * NFFT * PITCH;

  const int tid   = threadIdx.x;
  const int wave  = tid >> 5;
  const int lane  = tid & 31;
  const int lhalf = (lane >> 4);           // 0: lanes 0-15, 1: lanes 16-31

  // ---- Load this wave's 16 rows of x as A fragments (hi/lo f16), K = 256 ----
  // ISA 16-bit A 16x32 layout: lanes 0-15 row M=lane, halves 0-7 -> K 0..7,
  // halves 8-15 -> K 16..23 ; lanes 16-31 same row, K +8 / +24.
  const int grow   = blockIdx.x * ROWS_PER_BLOCK + wave * ROWS_PER_WAVE + (lane & 15);
  const int grow_c = grow < batch ? grow : (batch - 1);   // clamp: keep EXEC full
  const float* __restrict__ xrow = x + (size_t)grow_c * NFFT;

  v16h a_hi[KSTEPS];
  v16h a_lo[KSTEPS];
#pragma unroll
  for (int ks = 0; ks < KSTEPS; ++ks) {
    const int base = ks * 32 + (lhalf << 3);             // +0 or +8
    v4f p0 = *(const v4f*)(xrow + base);
    v4f p1 = *(const v4f*)(xrow + base + 4);
    v4f p2 = *(const v4f*)(xrow + base + 16);
    v4f p3 = *(const v4f*)(xrow + base + 20);
    FragA uh, ul;
#pragma unroll
    for (int i = 0; i < 4; ++i) {
      _Float16 h;
      h = (_Float16)p0[i]; uh.e[i]      = h; ul.e[i]      = (_Float16)(p0[i] - (float)h);
      h = (_Float16)p1[i]; uh.e[4 + i]  = h; ul.e[4 + i]  = (_Float16)(p1[i] - (float)h);
      h = (_Float16)p2[i]; uh.e[8 + i]  = h; ul.e[8 + i]  = (_Float16)(p2[i] - (float)h);
      h = (_Float16)p3[i]; uh.e[12 + i] = h; ul.e[12 + i] = (_Float16)(p3[i] - (float)h);
    }
    a_hi[ks] = uh.v;
    a_lo[ks] = ul.v;
  }

  const int out_col_lane = (lane & 15);
  const int out_row_base = blockIdx.x * ROWS_PER_BLOCK + wave * ROWS_PER_WAVE + (lhalf << 3);

  // ---- Loop over 4 column-quarters of the output (64 cols each) ----
  for (int q = 0; q < 4; ++q) {
    __syncthreads();   // previous quarter's B reads done before refill

    // Stage W_re/W_im columns [q*64, q*64+64) into LDS as f16 hi/lo.
    // B[k][n] = W[n][k]; the DFT matrix is symmetric so W[n][k] == W[k][n],
    // letting us read rows contiguously. 256*64/4 = 4096 float4s, 16/thread.
#pragma unroll 4
    for (int it = 0; it < 16; ++it) {
      const int idx = it * 256 + tid;       // 0..4095
      const int k   = idx >> 4;             // 16 float4s per row of 64
      const int n4  = (idx & 15) << 2;      // local col, multiple of 4
      const int n   = q * NQUART + n4;
      v4f r  = *(const v4f*)(dreal + k * NFFT + n);
      v4f im = *(const v4f*)(dimag + k * NFFT + n);
      v4h rh, rl, ih, il;
#pragma unroll
      for (int c = 0; c < 4; ++c) {
        _Float16 h;
        h = (_Float16)r[c];  rh[c] = h; rl[c] = (_Float16)(r[c]  - (float)h);
        h = (_Float16)im[c]; ih[c] = h; il[c] = (_Float16)(im[c] - (float)h);
      }
      const int lo = k * PITCH + n4;
      *(v4h*)(wr_hi + lo) = rh;
      *(v4h*)(wr_lo + lo) = rl;
      *(v4h*)(wi_hi + lo) = ih;
      *(v4h*)(wi_lo + lo) = il;
    }
    __syncthreads();

    // ---- 4 subtiles of 16 columns each within this quarter ----
#pragma unroll
    for (int st = 0; st < 4; ++st) {
      const int ncol = st * 16;
      v8f accr = {0.f, 0.f, 0.f, 0.f, 0.f, 0.f, 0.f, 0.f};
      v8f acci = {0.f, 0.f, 0.f, 0.f, 0.f, 0.f, 0.f, 0.f};

      // Ping-pong B fragments: prefetch k-step ks+1 while computing ks.
      FragB fr[2][4];
      load_frags(fr[0], wr_hi, wr_lo, wi_hi, wi_lo, (lhalf << 3) * PITCH + ncol + (lane & 15));

#pragma unroll
      for (int ks = 0; ks < KSTEPS; ++ks) {
        if (ks + 1 < KSTEPS) {
          const int lo_next = ((ks + 1) * 32 + (lhalf << 3)) * PITCH + ncol + (lane & 15);
          load_frags(fr[(ks + 1) & 1], wr_hi, wr_lo, wi_hi, wi_lo, lo_next);
        }
        const FragB* f = fr[ks & 1];
        // Interleave the two independent accumulation chains (re / im).
        accr = wmma_f16(a_hi[ks], f[0].v, accr);
        acci = wmma_f16(a_hi[ks], f[2].v, acci);
        accr = wmma_f16(a_lo[ks], f[0].v, accr);
        acci = wmma_f16(a_lo[ks], f[2].v, acci);
        accr = wmma_f16(a_hi[ks], f[1].v, accr);
        acci = wmma_f16(a_hi[ks], f[3].v, acci);
      }

      // Epilogue: log(re^2 + im^2). C/D layout: VGPR v -> row v (lanes 0-15)
      // or v+8 (lanes 16-31), col = lane&15.
      const int col = q * NQUART + ncol + out_col_lane;
#pragma unroll
      for (int v = 0; v < 8; ++v) {
        const int r = out_row_base + v;
        if (r < batch) {
          out[(size_t)r * NFFT + col] = __logf(accr[v] * accr[v] + acci[v] * acci[v]);
        }
      }
    }
  }
}

extern "C" void kernel_launch(void* const* d_in, const int* in_sizes, int n_in,
                              void* d_out, int out_size, void* d_ws, size_t ws_size,
                              hipStream_t stream) {
  const float* x     = (const float*)d_in[0];
  const float* dreal = (const float*)d_in[1];
  const float* dimag = (const float*)d_in[2];
  float* out = (float*)d_out;

  const int batch  = in_sizes[0] / NFFT;
  const int blocks = (batch + ROWS_PER_BLOCK - 1) / ROWS_PER_BLOCK;
  const size_t lds_bytes = (size_t)4 * NFFT * PITCH * sizeof(_Float16);  // 147456

  dft_logmag_kernel<<<blocks, WAVES * 32, lds_bytes, stream>>>(x, dreal, dimag, out, batch);
}
